// TransformerBlock_72722386255908
// MI455X (gfx1250) — matrix operand order repaired and checked by reading, not compiled, not run on hardware
//
#include <hip/hip_runtime.h>
#include <hip/hip_bf16.h>

typedef __bf16 bf16;
typedef __attribute__((ext_vector_type(16))) __bf16 v16bf;
typedef __attribute__((ext_vector_type(8)))  float  v8f;

#define TOK     8192   // B*S
#define CDIM    1024
#define SEQ     2048
#define BATCH   4
#define NHEADS  16
#define KVHEADS 4
#define HDIM    64
#define KVC     256    // KVHEADS*HDIM
#define NEXP    8

static __device__ __forceinline__ v8f zero8() {
  v8f z;
#pragma unroll
  for (int i = 0; i < 8; i++) z[i] = 0.0f;
  return z;
}

// Build a v16bf fragment from two 16B chunks (LDS or global).
static __device__ __forceinline__ v16bf ldfrag(const bf16* p0, const bf16* p1) {
  union { uint4 u[2]; v16bf v; } r;
  r.u[0] = *(const uint4*)p0;
  r.u[1] = *(const uint4*)p1;
  return r.v;
}

// DPP16 lane permute (pure VALU, no LDS round-trip). CTRL must be immediate.
template <int CTRL>
static __device__ __forceinline__ float dppf(float v) {
  return __int_as_float(__builtin_amdgcn_update_dpp(
      __float_as_int(v), __float_as_int(v), CTRL, 0xf, 0xf, true));
}
#define DPP_QX1 0xB1   // quad_perm [1,0,3,2]
#define DPP_QX2 0x4E   // quad_perm [2,3,0,1]
#define DPP_RHM 0x141  // row_half_mirror
#define DPP_RM  0x140  // row_mirror

// Butterfly reductions over each 16-lane row (all 16 lanes end with result).
static __device__ __forceinline__ float row_max16(float v) {
  v = fmaxf(v, dppf<DPP_QX1>(v));
  v = fmaxf(v, dppf<DPP_QX2>(v));
  v = fmaxf(v, dppf<DPP_RHM>(v));
  v = fmaxf(v, dppf<DPP_RM>(v));
  return v;
}
static __device__ __forceinline__ float row_sum16(float v) {
  v += dppf<DPP_QX1>(v);
  v += dppf<DPP_QX2>(v);
  v += dppf<DPP_RHM>(v);
  v += dppf<DPP_RM>(v);
  return v;
}

// ---------------------------------------------------------------------------
// RMSNorm (fp32 stats) -> bf16 output.  One block per token, 256 threads.
// ---------------------------------------------------------------------------
__global__ __launch_bounds__(256)
void rmsnorm_k(const float* __restrict__ x, const float* __restrict__ w,
               bf16* __restrict__ out) {
  const int t = blockIdx.x;
  const int tid = threadIdx.x;
  const float4 xv = ((const float4*)(x + (size_t)t * CDIM))[tid];
  float ss = xv.x * xv.x + xv.y * xv.y + xv.z * xv.z + xv.w * xv.w;
#pragma unroll
  for (int m = 1; m < 32; m <<= 1) ss += __shfl_xor(ss, m);
  __shared__ float wsum[8];
  if ((tid & 31) == 0) wsum[tid >> 5] = ss;
  __syncthreads();
  float tot = 0.f;
#pragma unroll
  for (int i = 0; i < 8; i++) tot += wsum[i];
  const float rs = rsqrtf(tot * (1.0f / CDIM) + 1e-6f);
  const float4 wv = ((const float4*)w)[tid];
  bf16* o = out + (size_t)t * CDIM + tid * 4;
  o[0] = (bf16)(xv.x * rs * wv.x);
  o[1] = (bf16)(xv.y * rs * wv.y);
  o[2] = (bf16)(xv.z * rs * wv.z);
  o[3] = (bf16)(xv.w * rs * wv.w);
}

// ---------------------------------------------------------------------------
// Tiled WMMA GEMM:  out[M,N] = A_bf16[M,K] @ W_f32[N,K]^T  (K = 1024)
// Block tile 128x128, K-step 32, 256 threads = 8 waves (4M x 2N),
// double-buffered LDS (one barrier per K-step, loads prefetched one step).
// MODE 0: outF = acc + residual         (wo projection -> h)
// MODE 1: outB = bf16(acc)              (Q/K/V projections)
// ---------------------------------------------------------------------------
#define GLDP 40  // padded LDS row pitch in bf16 elements (80B, 16B-aligned)

template <int MODE>
__global__ __launch_bounds__(256)
void gemm_k(const bf16* __restrict__ A, const float* __restrict__ W, int ldN,
            float* __restrict__ outF, bf16* __restrict__ outB,
            const float* __restrict__ residual) {
  __shared__ __align__(16) bf16 Asl[2][128 * GLDP];
  __shared__ __align__(16) bf16 Wsl[2][128 * GLDP];
  const int tid  = threadIdx.x;
  const int lane = tid & 31;
  const int wave = tid >> 5;
  const int mw = wave >> 1;       // 0..3
  const int nw = wave & 1;        // 0..1
  const int row0 = blockIdx.y * 128;
  const int col0 = blockIdx.x * 128;
  const int r  = lane & 15;
  const int hh = lane >> 4;

  v8f acc[2][4];
#pragma unroll
  for (int ms = 0; ms < 2; ms++)
#pragma unroll
    for (int ns = 0; ns < 4; ns++) acc[ms][ns] = zero8();

  const int lrow = tid >> 1;          // 0..127
  const int lcol = (tid & 1) * 16;    // 0 or 16
  const bf16*  Ag = A + (size_t)(row0 + lrow) * CDIM + lcol;
  const float* Wg = W + (size_t)(col0 + lrow) * CDIM + lcol;
  const int soff = lrow * GLDP + lcol;

  uint4 pa0, pa1;
  union Pk { bf16 h[16]; uint4 u[2]; } pw;

  auto gload = [&](int k0) {
    pa0 = ((const uint4*)(Ag + k0))[0];
    pa1 = ((const uint4*)(Ag + k0))[1];
    const float4 f0 = ((const float4*)(Wg + k0))[0];
    const float4 f1 = ((const float4*)(Wg + k0))[1];
    const float4 f2 = ((const float4*)(Wg + k0))[2];
    const float4 f3 = ((const float4*)(Wg + k0))[3];
    pw.h[0]  = (bf16)f0.x; pw.h[1]  = (bf16)f0.y; pw.h[2]  = (bf16)f0.z; pw.h[3]  = (bf16)f0.w;
    pw.h[4]  = (bf16)f1.x; pw.h[5]  = (bf16)f1.y; pw.h[6]  = (bf16)f1.z; pw.h[7]  = (bf16)f1.w;
    pw.h[8]  = (bf16)f2.x; pw.h[9]  = (bf16)f2.y; pw.h[10] = (bf16)f2.z; pw.h[11] = (bf16)f2.w;
    pw.h[12] = (bf16)f3.x; pw.h[13] = (bf16)f3.y; pw.h[14] = (bf16)f3.z; pw.h[15] = (bf16)f3.w;
  };
  auto sstore = [&](int buf) {
    ((uint4*)(Asl[buf] + soff))[0] = pa0;
    ((uint4*)(Asl[buf] + soff))[1] = pa1;
    ((uint4*)(Wsl[buf] + soff))[0] = pw.u[0];
    ((uint4*)(Wsl[buf] + soff))[1] = pw.u[1];
  };

  gload(0);
  sstore(0);
  __syncthreads();

  const int NK = CDIM / 32;
  for (int ki = 0; ki < NK; ki++) {
    const int cur = ki & 1;
    if (ki + 1 < NK) gload((ki + 1) * 32);   // prefetch next step

    v16bf af[2];
#pragma unroll
    for (int ms = 0; ms < 2; ms++) {
      const bf16* p = Asl[cur] + (mw * 32 + ms * 16 + r) * GLDP + hh * 8;
      af[ms] = ldfrag(p, p + 16);
    }
#pragma unroll
    for (int ns = 0; ns < 4; ns++) {
      const bf16* p = Wsl[cur] + (nw * 64 + ns * 16 + r) * GLDP + hh * 8;
      v16bf bfr = ldfrag(p, p + 16);
#pragma unroll
      for (int ms = 0; ms < 2; ms++)
        acc[ms][ns] = __builtin_amdgcn_wmma_f32_16x16x32_bf16(
            false, af[ms], false, bfr, (short)0, acc[ms][ns], false, false);
    }
    if (ki + 1 < NK) sstore(1 - cur);        // write the idle buffer
    __syncthreads();
  }

#pragma unroll
  for (int ms = 0; ms < 2; ms++) {
#pragma unroll
    for (int ns = 0; ns < 4; ns++) {
      const int col = col0 + nw * 64 + ns * 16 + r;
#pragma unroll
      for (int v = 0; v < 8; v++) {
        const int row = row0 + mw * 32 + ms * 16 + v + 8 * hh;
        const size_t idx = (size_t)row * ldN + col;
        const float val = acc[ms][ns][v];
        if (MODE == 0)      outF[idx] = val + residual[idx];
        else                outB[idx] = (bf16)val;
      }
    }
  }
}

// ---------------------------------------------------------------------------
// Routed MoE gather-GEMM.  grid.z = expert.  Each block handles 128 gathered
// token rows of expert e (early-exit past cnt[e]); scatter-accumulates
// out[token] += w_token * (hn[token] @ We^T).  Same WMMA core as gemm_k.
// ---------------------------------------------------------------------------
__global__ __launch_bounds__(256)
void moe_gemm_k(const bf16* __restrict__ A, const float* __restrict__ Wbase,
                const int* __restrict__ cnt, const int* __restrict__ list,
                const float* __restrict__ wgt, float* __restrict__ outF) {
  const int e = blockIdx.z;
  const int M = cnt[e];
  const int row0 = blockIdx.y * 128;
  if (row0 >= M) return;
  const float* W = Wbase + (size_t)e * CDIM * CDIM;
  const int* lst = list + e * TOK;
  const float* wg = wgt + e * TOK;

  __shared__ __align__(16) bf16 Asl[2][128 * GLDP];
  __shared__ __align__(16) bf16 Wsl[2][128 * GLDP];
  __shared__ int   Lsl[128];   // gathered token ids for this tile
  __shared__ float Fsl[128];   // per-token routing weights

  const int tid  = threadIdx.x;
  const int lane = tid & 31;
  const int wave = tid >> 5;
  const int mw = wave >> 1;
  const int nw = wave & 1;
  const int col0 = blockIdx.x * 128;
  const int r  = lane & 15;
  const int hh = lane >> 4;

  if (tid < 128) {
    int gi = row0 + tid;
    const int gc = (gi < M) ? gi : (M - 1);
    Lsl[tid] = lst[gc];
    Fsl[tid] = (gi < M) ? wg[gi] : 0.0f;
  }
  __syncthreads();

  v8f acc[2][4];
#pragma unroll
  for (int ms = 0; ms < 2; ms++)
#pragma unroll
    for (int ns = 0; ns < 4; ns++) acc[ms][ns] = zero8();

  const int lrow = tid >> 1;
  const int lcol = (tid & 1) * 16;
  const bf16*  Ag = A + (size_t)Lsl[lrow] * CDIM + lcol;   // gathered A row
  const float* Wg = W + (size_t)(col0 + lrow) * CDIM + lcol;
  const int soff = lrow * GLDP + lcol;

  uint4 pa0, pa1;
  union Pk { bf16 h[16]; uint4 u[2]; } pw;

  auto gload = [&](int k0) {
    pa0 = ((const uint4*)(Ag + k0))[0];
    pa1 = ((const uint4*)(Ag + k0))[1];
    const float4 f0 = ((const float4*)(Wg + k0))[0];
    const float4 f1 = ((const float4*)(Wg + k0))[1];
    const float4 f2 = ((const float4*)(Wg + k0))[2];
    const float4 f3 = ((const float4*)(Wg + k0))[3];
    pw.h[0]  = (bf16)f0.x; pw.h[1]  = (bf16)f0.y; pw.h[2]  = (bf16)f0.z; pw.h[3]  = (bf16)f0.w;
    pw.h[4]  = (bf16)f1.x; pw.h[5]  = (bf16)f1.y; pw.h[6]  = (bf16)f1.z; pw.h[7]  = (bf16)f1.w;
    pw.h[8]  = (bf16)f2.x; pw.h[9]  = (bf16)f2.y; pw.h[10] = (bf16)f2.z; pw.h[11] = (bf16)f2.w;
    pw.h[12] = (bf16)f3.x; pw.h[13] = (bf16)f3.y; pw.h[14] = (bf16)f3.z; pw.h[15] = (bf16)f3.w;
  };
  auto sstore = [&](int buf) {
    ((uint4*)(Asl[buf] + soff))[0] = pa0;
    ((uint4*)(Asl[buf] + soff))[1] = pa1;
    ((uint4*)(Wsl[buf] + soff))[0] = pw.u[0];
    ((uint4*)(Wsl[buf] + soff))[1] = pw.u[1];
  };

  gload(0);
  sstore(0);
  __syncthreads();

  const int NK = CDIM / 32;
  for (int ki = 0; ki < NK; ki++) {
    const int cur = ki & 1;
    if (ki + 1 < NK) gload((ki + 1) * 32);
    v16bf af[2];
#pragma unroll
    for (int ms = 0; ms < 2; ms++) {
      const bf16* p = Asl[cur] + (mw * 32 + ms * 16 + r) * GLDP + hh * 8;
      af[ms] = ldfrag(p, p + 16);
    }
#pragma unroll
    for (int ns = 0; ns < 4; ns++) {
      const bf16* p = Wsl[cur] + (nw * 64 + ns * 16 + r) * GLDP + hh * 8;
      v16bf bfr = ldfrag(p, p + 16);
#pragma unroll
      for (int ms = 0; ms < 2; ms++)
        acc[ms][ns] = __builtin_amdgcn_wmma_f32_16x16x32_bf16(
            false, af[ms], false, bfr, (short)0, acc[ms][ns], false, false);
    }
    if (ki + 1 < NK) sstore(1 - cur);
    __syncthreads();
  }

#pragma unroll
  for (int ms = 0; ms < 2; ms++) {
#pragma unroll
    for (int ns = 0; ns < 4; ns++) {
      const int col = col0 + nw * 64 + ns * 16 + r;
#pragma unroll
      for (int v = 0; v < 8; v++) {
        const int rl = mw * 32 + ms * 16 + v + 8 * hh;   // 0..127 in tile
        if (row0 + rl < M) {
          const size_t idx = (size_t)Lsl[rl] * CDIM + col;
          outF[idx] += Fsl[rl] * acc[ms][ns][v];
        }
      }
    }
  }
}

// ---------------------------------------------------------------------------
// Flash attention: one block per (q-tile of 64, head, batch). 128 thr = 4 waves,
// each wave owns 16 query rows. Online softmax in fp32 with DPP16 row
// reductions, QK^T and PV via WMMA.
// Output (pre-wo, bf16) may alias Q: each block touches only its own slice.
// ---------------------------------------------------------------------------
__global__ __launch_bounds__(128)
void attn_k(const bf16* Q, const bf16* __restrict__ Kg,
            const bf16* __restrict__ Vg, bf16* O) {
  __shared__ __align__(16) bf16 Ksl[64 * 72];        // [kv][d]
  __shared__ __align__(16) bf16 Vtsl[64 * 72];       // [d][kv]
  __shared__ __align__(16) bf16 Psl[4][16 * 72];     // per wave [q][kv]

  const int qt   = blockIdx.x;
  const int head = blockIdx.y;
  const int b    = blockIdx.z;
  const int kvh  = head >> 2;   // n_rep = 4
  const int q0   = qt * 64;
  const int tid  = threadIdx.x;
  const int lane = tid & 31;
  const int wave = tid >> 5;
  const int r  = lane & 15;
  const int hh = lane >> 4;
  bf16* Pw = &Psl[wave][0];

  const size_t qoff = (size_t)(b * SEQ + q0 + wave * 16 + r) * CDIM + head * HDIM;
  v16bf qf[2];
#pragma unroll
  for (int ks = 0; ks < 2; ks++) {
    const bf16* pq = Q + qoff + ks * 32 + hh * 8;
    qf[ks] = ldfrag(pq, pq + 16);
  }

  float mrow[8], lrow[8];
  v8f o_acc[4];
#pragma unroll
  for (int v = 0; v < 8; v++) { mrow[v] = -__builtin_inff(); lrow[v] = 0.f; }
#pragma unroll
  for (int nt = 0; nt < 4; nt++) o_acc[nt] = zero8();

  const int kvrow = tid >> 1;
  const int kvseg = (tid & 1) * 32;

  for (int jt = 0; jt <= qt; jt++) {     // causal: only tiles at/below diagonal
    const int j0 = jt * 64;
    __syncthreads();
    {  // stage K tile row-major (vectorized), V tile transposed
      const uint4* src =
          (const uint4*)(Kg + (size_t)(b * SEQ + j0 + kvrow) * KVC + kvh * HDIM + kvseg);
      uint4* dst = (uint4*)(Ksl + kvrow * 72 + kvseg);
      dst[0] = src[0]; dst[1] = src[1]; dst[2] = src[2]; dst[3] = src[3];
#pragma unroll
      for (int cc = 0; cc < 4; cc++) {
        const int e8 = cc * 128 + tid;          // 512 chunks of 8 halves
        const int vkv = e8 >> 3;
        const int dbase = (e8 & 7) * 8;
        union { uint4 u; bf16 h[8]; } vv;
        vv.u = *(const uint4*)(Vg + (size_t)(b * SEQ + j0 + vkv) * KVC +
                               kvh * HDIM + dbase);
#pragma unroll
        for (int j = 0; j < 8; j++) Vtsl[(dbase + j) * 72 + vkv] = vv.h[j];
      }
    }
    __syncthreads();

    v8f s[4];
#pragma unroll
    for (int nt = 0; nt < 4; nt++) {
      s[nt] = zero8();
#pragma unroll
      for (int ks = 0; ks < 2; ks++) {
        const bf16* pk = Ksl + (nt * 16 + r) * 72 + ks * 32 + hh * 8;
        v16bf kf = ldfrag(pk, pk + 16);
        s[nt] = __builtin_amdgcn_wmma_f32_16x16x32_bf16(
            false, qf[ks], false, kf, (short)0, s[nt], false, false);
      }
    }

    const bool diag = (jt == qt);
#pragma unroll
    for (int nt = 0; nt < 4; nt++) {
#pragma unroll
      for (int v = 0; v < 8; v++) {
        float sv = s[nt][v] * 0.125f;  // 1/sqrt(64)
        if (diag && (nt * 16 + r) > (wave * 16 + v + 8 * hh)) sv = -__builtin_inff();
        s[nt][v] = sv;
      }
    }

#pragma unroll
    for (int v = 0; v < 8; v++) {
      float rm = fmaxf(fmaxf(s[0][v], s[1][v]), fmaxf(s[2][v], s[3][v]));
      rm = row_max16(rm);
      const float mn = fmaxf(mrow[v], rm);
      const float fv = __expf(mrow[v] - mn);
      mrow[v] = mn;
      float ps = 0.f;
#pragma unroll
      for (int nt = 0; nt < 4; nt++) {
        const float pe = __expf(s[nt][v] - mn);
        s[nt][v] = pe;
        ps += pe;
      }
      ps = row_sum16(ps);
      lrow[v] = lrow[v] * fv + ps;
#pragma unroll
      for (int nt = 0; nt < 4; nt++) o_acc[nt][v] *= fv;
      const int prow = (v + 8 * hh) * 72;
#pragma unroll
      for (int nt = 0; nt < 4; nt++) Pw[prow + nt * 16 + r] = (bf16)s[nt][v];
    }

    v16bf pf[2];
#pragma unroll
    for (int ks = 0; ks < 2; ks++) {
      const bf16* pp = Pw + r * 72 + ks * 32 + hh * 8;
      pf[ks] = ldfrag(pp, pp + 16);
    }
#pragma unroll
    for (int nt = 0; nt < 4; nt++) {
#pragma unroll
      for (int ks = 0; ks < 2; ks++) {
        const bf16* pv = Vtsl + (nt * 16 + r) * 72 + ks * 32 + hh * 8;
        v16bf vf = ldfrag(pv, pv + 16);
        o_acc[nt] = __builtin_amdgcn_wmma_f32_16x16x32_bf16(
            false, pf[ks], false, vf, (short)0, o_acc[nt], false, false);
      }
    }
  }

#pragma unroll
  for (int v = 0; v < 8; v++) {
    const float inv = 1.0f / lrow[v];
    const size_t orow =
        (size_t)(b * SEQ + q0 + wave * 16 + v + 8 * hh) * CDIM + head * HDIM;
#pragma unroll
    for (int nt = 0; nt < 4; nt++)
      O[orow + nt * 16 + r] = (bf16)(o_acc[nt][v] * inv);
  }
}

// ---------------------------------------------------------------------------
// Gate + routing: logits = hn @ gate_w^T, top-2 (ties -> lowest index, like
// jax.lax.top_k), softmax over the pair, append token to the two experts'
// gather lists.  One wave per token.
// ---------------------------------------------------------------------------
__global__ __launch_bounds__(256)
void gate_k(const bf16* __restrict__ hn, const float* __restrict__ gw,
            int* __restrict__ cnt, int* __restrict__ list,
            float* __restrict__ wgt) {
  const int lane = threadIdx.x & 31;
  const int wave = threadIdx.x >> 5;
  const int t = blockIdx.x * 8 + wave;
  const bf16* hr = hn + (size_t)t * CDIM;
  float logits[NEXP];
#pragma unroll
  for (int e = 0; e < NEXP; e++) {
    const float* g = gw + e * CDIM;
    float s = 0.f;
    for (int i = lane; i < CDIM; i += 32) s += (float)hr[i] * g[i];
#pragma unroll
    for (int m = 1; m < 32; m <<= 1) s += __shfl_xor(s, m);
    logits[e] = s;
  }
  int e0 = 0; float l0 = logits[0];
#pragma unroll
  for (int e = 1; e < NEXP; e++) if (logits[e] > l0) { l0 = logits[e]; e0 = e; }
  int e1 = -1; float l1 = -__builtin_inff();
#pragma unroll
  for (int e = 0; e < NEXP; e++)
    if (e != e0 && logits[e] > l1) { l1 = logits[e]; e1 = e; }
  const float w1e = __expf(l1 - l0);
  const float inv = 1.0f / (1.0f + w1e);
  if (lane < 2) {
    const int   me = (lane == 0) ? e0 : e1;
    const float mw = (lane == 0) ? inv : w1e * inv;
    const int pos = atomicAdd(&cnt[me], 1);
    list[me * TOK + pos] = t;
    wgt[me * TOK + pos] = mw;
  }
}

__global__ void zero_cnt_k(int* __restrict__ cnt) {
  if (threadIdx.x < NEXP) cnt[threadIdx.x] = 0;
}

__global__ __launch_bounds__(256)
void copy_k(const float* __restrict__ in, float* __restrict__ o) {
  const size_t i = (size_t)blockIdx.x * 256 + threadIdx.x;
  ((float4*)o)[i] = ((const float4*)in)[i];
}

// ---------------------------------------------------------------------------
extern "C" void kernel_launch(void* const* d_in, const int* in_sizes, int n_in,
                              void* d_out, int out_size, void* d_ws,
                              size_t ws_size, hipStream_t stream) {
  (void)in_sizes; (void)n_in; (void)out_size; (void)ws_size;
  const float* x   = (const float*)d_in[0];
  const float* anw = (const float*)d_in[1];
  const float* fnw = (const float*)d_in[2];
  const float* wq  = (const float*)d_in[3];
  const float* wk  = (const float*)d_in[4];
  const float* wv  = (const float*)d_in[5];
  const float* wo  = (const float*)d_in[6];
  const float* gw  = (const float*)d_in[7];
  const float* ew  = (const float*)d_in[8];
  float* out = (float*)d_out;

  char* p = (char*)d_ws;
  bf16* xn = (bf16*)p;    p += (size_t)TOK * CDIM * 2;  // normed acts (reused as hn)
  bf16* qb = (bf16*)p;    p += (size_t)TOK * CDIM * 2;  // Q, reused as attn output
  bf16* kb = (bf16*)p;    p += (size_t)TOK * KVC * 2;
  bf16* vb = (bf16*)p;    p += (size_t)TOK * KVC * 2;
  float* hbuf = (float*)p; p += (size_t)TOK * CDIM * 4; // residual stream h
  int*   cnt  = (int*)p;   p += 256;                    // 8 counters (padded)
  int*   list = (int*)p;   p += (size_t)NEXP * TOK * 4; // gather lists
  float* wgt  = (float*)p; p += (size_t)NEXP * TOK * 4; // routing weights

  // attention branch
  rmsnorm_k<<<TOK, 256, 0, stream>>>(x, anw, xn);
  gemm_k<1><<<dim3(CDIM / 128, TOK / 128), 256, 0, stream>>>(
      xn, wq, CDIM, nullptr, qb, nullptr);
  gemm_k<1><<<dim3(KVC / 128, TOK / 128), 256, 0, stream>>>(
      xn, wk, KVC, nullptr, kb, nullptr);
  gemm_k<1><<<dim3(KVC / 128, TOK / 128), 256, 0, stream>>>(
      xn, wv, KVC, nullptr, vb, nullptr);
  attn_k<<<dim3(SEQ / 64, NHEADS, BATCH), 128, 0, stream>>>(qb, kb, vb, qb);
  gemm_k<0><<<dim3(CDIM / 128, TOK / 128), 256, 0, stream>>>(
      qb, wo, CDIM, hbuf, nullptr, x);                   // h = x + attn@wo^T

  // MoE branch (routed top-2: ~4x less GEMM work than dense 8-expert loop)
  rmsnorm_k<<<TOK, 256, 0, stream>>>(hbuf, fnw, xn);     // hn (reuse xn)
  zero_cnt_k<<<1, 32, 0, stream>>>(cnt);
  gate_k<<<TOK / 8, 256, 0, stream>>>(xn, gw, cnt, list, wgt);
  copy_k<<<(TOK * CDIM / 4) / 256, 256, 0, stream>>>(hbuf, out);  // out = h
  moe_gemm_k<<<dim3(CDIM / 128, TOK / 128, NEXP), 256, 0, stream>>>(
      xn, ew, cnt, list, wgt, out);
}
